// QConv2d_69587060130170
// MI455X (gfx1250) — 8-bit matrix operand order repaired and checked by reading, compiled, not run on hardware
//
#include <hip/hip_runtime.h>
#include <stdint.h>

typedef __attribute__((ext_vector_type(8))) int v8i;

#define CIN      256
#define COUT     256
#define HH       56
#define WW       56
#define HP       58
#define WP       58
#define BATCH    32
#define SPATIAL  (HH*WW)
#define KTOT     (CIN*9)
#define KCHUNK   64
#define NCHUNK   (KTOT/KCHUNK)
#define TILE_M   64
#define TILE_N   256
#define A_STRIDE 80
#define B_STRIDE 80
#define A_BYTES  (TILE_M*A_STRIDE)
#define B_BYTES  (TILE_N*B_STRIDE)
#define BUF_BYTES (A_BYTES + B_BYTES)
#define IXQ_BYTES ((size_t)BATCH*HP*WP*CIN)

__device__ __forceinline__ void async_b128(unsigned lds_off, const void* sbase, unsigned voff) {
  asm volatile("global_load_async_to_lds_b128 %0, %1, %2"
               :: "v"(lds_off), "v"(voff), "s"(sbase) : "memory");
}
__device__ __forceinline__ void wait_async0() {
  asm volatile("s_wait_asynccnt 0x0" ::: "memory");
}

__global__ __launch_bounds__(1024) void quant_x_kernel(
    const float* __restrict__ x, const float* __restrict__ sxp,
    signed char* __restrict__ ixq)
{
  __shared__ signed char tile[32][33];
  const float sx = sxp[0];
  const int tx = threadIdx.x, ty = threadIdx.y;
  const int w0 = blockIdx.x * 32;
  const int c0 = blockIdx.y * 32;
  const int bh = blockIdx.z;
  const int b  = bh / HH, h = bh % HH;

  int w = w0 + tx;
  int c = c0 + ty;
  signed char q = 0;
  if (w < WW) {
    float v = x[((b*CIN + c)*HH + h)*WW + w] * sx;
    v = fminf(fmaxf(v, -128.0f), 127.0f);
    q = (signed char)(int)rintf(v);
  }
  tile[ty][tx] = q;
  __syncthreads();
  int w1 = w0 + ty;
  int c1 = c0 + tx;
  if (w1 < WW)
    ixq[((b*HP + h + 1)*WP + w1 + 1)*CIN + c1] = tile[tx][ty];
}

__global__ __launch_bounds__(256) void quant_w_kernel(
    const float* __restrict__ wgt, const float* __restrict__ sw,
    unsigned int* __restrict__ qw)
{
  int tid = blockIdx.x * 256 + threadIdx.x;
  int co  = tid / (KTOT/4);
  int g   = tid % (KTOT/4);
  int khw = g >> 6;
  int ci  = (g & 63) * 4;
  float s = sw[co];
  unsigned int pack = 0;
#pragma unroll
  for (int j = 0; j < 4; ++j) {
    float v = wgt[(co*CIN + ci + j)*9 + khw] * s;
    v = fminf(fmaxf(v, -128.0f), 127.0f);
    int q = (int)rintf(v);
    pack |= ((unsigned int)(q & 0xff)) << (8*j);
  }
  qw[co*(KTOT/4) + g] = pack;
}

__global__ __launch_bounds__(256) void qconv_wmma_kernel(
    const signed char*   __restrict__ ixq,
    const unsigned char* __restrict__ qwb,
    const float* __restrict__ bias,
    const float* __restrict__ sw,
    const float* __restrict__ sxp,
    float* __restrict__ out)
{
  __shared__ __align__(16) signed char lds[2*BUF_BYTES];
  const unsigned lds0 = (unsigned)(uintptr_t)(&lds[0]);

  const int t    = threadIdx.x;
  const int wave = t >> 5;
  const int lane = t & 31;
  const int wm   = wave & 1;
  const int wn   = wave >> 1;

  const int m_base = blockIdx.x * TILE_M;
  const int batch  = m_base / SPATIAL;
  const int s_base = m_base - batch * SPATIAL;

  const int col16 = t & 3;
  const int rowA  = t >> 2;
  const int coB   = t >> 2;
  {
  }
  int sA = s_base + rowA;
  int hA = sA / WW;
  int wA = sA - hA*WW;
  const int arowbase = (batch*HP + hA)*WP + wA;

  const int lrow = lane & 15;
  const int kbA  = (lane >> 4) * 16;
  const int kbB  = (lane >> 4) * 16;

  v8i acc[2][4] = {};

  auto stage = [&](int chunk, int buf) {
    const int khw = chunk >> 2;
    const int kh  = khw / 3;
    const int kw  = khw - kh*3;
    const int c0  = (chunk & 3) * KCHUNK;
    const unsigned bufoff = lds0 + (unsigned)(buf * BUF_BYTES);
    unsigned aoff = (unsigned)((arowbase + kh*WP + kw)*CIN + c0 + col16*16);
    async_b128(bufoff + rowA*A_STRIDE + col16*16, ixq, aoff);
#pragma unroll
    for (int i = 0; i < 4; ++i) {
      int co = coB + i*64;
      unsigned boff = (unsigned)(co*KTOT + chunk*KCHUNK + col16*16);
      async_b128(bufoff + A_BYTES + co*B_STRIDE + col16*16, qwb, boff);
    }
  };

  stage(0, 0);
  wait_async0();
  __syncthreads();

  for (int chunk = 0; chunk < NCHUNK; ++chunk) {
    const int cur = chunk & 1;
    if (chunk + 1 < NCHUNK) stage(chunk + 1, cur ^ 1);

    const signed char* ldsA = lds + cur*BUF_BYTES;
    const signed char* ldsB = ldsA + A_BYTES;

    v8i afr[2], bfr[4];
#pragma unroll
    for (int mt = 0; mt < 2; ++mt) {
      const signed char* p = ldsA + ((wm*2 + mt)*16 + lrow)*A_STRIDE + kbA;
      int2 p0 = *(const int2*)(p +  0);
      int2 p1 = *(const int2*)(p +  8);
      int2 p2 = *(const int2*)(p + 32);
      int2 p3 = *(const int2*)(p + 40);
      v8i a; a[0]=p0.x; a[1]=p0.y; a[2]=p1.x; a[3]=p1.y;
             a[4]=p2.x; a[5]=p2.y; a[6]=p3.x; a[7]=p3.y;
      afr[mt] = a;
    }
#pragma unroll
    for (int nt = 0; nt < 4; ++nt) {
      const signed char* p = ldsB + ((wn*4 + nt)*16 + lrow)*B_STRIDE + kbB;
      int4 q0 = *(const int4*)(p);
      int4 q1 = *(const int4*)(p + 32);
      v8i b; b[0]=q0.x; b[1]=q0.y; b[2]=q0.z; b[3]=q0.w;
             b[4]=q1.x; b[5]=q1.y; b[6]=q1.z; b[7]=q1.w;
      bfr[nt] = b;
    }
#pragma unroll
    for (int mt = 0; mt < 2; ++mt)
#pragma unroll
      for (int nt = 0; nt < 4; ++nt)
        acc[mt][nt] = __builtin_amdgcn_wmma_i32_16x16x64_iu8(
            true, afr[mt], true, bfr[nt], acc[mt][nt], false, false);

    wait_async0();
    __syncthreads();
  }

  const float sx = sxp[0];
#pragma unroll
  for (int nt = 0; nt < 4; ++nt) {
    int co = (wn*4 + nt)*16 + lrow;
    float sc = 1.0f / (sw[co] * sx);
    float bi = bias[co];
#pragma unroll
    for (int mt = 0; mt < 2; ++mt) {
      long obase = (long)batch*(COUT*(long)SPATIAL) + (long)co*SPATIAL
                 + s_base + (wm*2 + mt)*16 + (lane >> 4)*8;
      float4 r0, r1;
      r0.x = (float)acc[mt][nt][0]*sc + bi;
      r0.y = (float)acc[mt][nt][1]*sc + bi;
      r0.z = (float)acc[mt][nt][2]*sc + bi;
      r0.w = (float)acc[mt][nt][3]*sc + bi;
      r1.x = (float)acc[mt][nt][4]*sc + bi;
      r1.y = (float)acc[mt][nt][5]*sc + bi;
      r1.z = (float)acc[mt][nt][6]*sc + bi;
      r1.w = (float)acc[mt][nt][7]*sc + bi;
      *(float4*)(out + obase)     = r0;
      *(float4*)(out + obase + 4) = r1;
    }
  }
}

extern "C" void kernel_launch(void* const* d_in, const int* in_sizes, int n_in,
                              void* d_out, int out_size, void* d_ws, size_t ws_size,
                              hipStream_t stream) {
  (void)in_sizes; (void)n_in; (void)out_size; (void)ws_size;
  const float* x    = (const float*)d_in[0];
  const float* wgt  = (const float*)d_in[1];
  const float* bias = (const float*)d_in[2];
  const float* sw   = (const float*)d_in[3];
  const float* sx   = (const float*)d_in[4];
  float* out = (float*)d_out;

  signed char*  ixq = (signed char*)d_ws;
  unsigned int* qw  = (unsigned int*)((char*)d_ws + IXQ_BYTES);

  hipMemsetAsync(ixq, 0, IXQ_BYTES, stream);
  quant_x_kernel<<<dim3(2, CIN/32, BATCH*HH), dim3(32, 32), 0, stream>>>(x, sx, ixq);
  quant_w_kernel<<<dim3((COUT*(KTOT/4))/256), dim3(256), 0, stream>>>(wgt, sw, qw);
  qconv_wmma_kernel<<<dim3((BATCH*SPATIAL)/TILE_M), dim3(256), 0, stream>>>(
      ixq, (const unsigned char*)qw, bias, sw, sx, out);
}
